// MoEAudioProjector_18451179504411
// MI455X (gfx1250) — matrix operand order repaired and checked by reading, not compiled, not run on hardware
//
#include <hip/hip_runtime.h>
#include <hip/hip_bf16.h>

// ---------------- Problem constants ----------------
#define T_TOK   12000          // 16 * (1500/2) tokens
#define IN_DIM  2560
#define HID     512
#define HID2    1024           // 2*HID
#define LLM     2048
#define NEXP    8

typedef _Float16 h16;
typedef __attribute__((ext_vector_type(16))) _Float16 v16h;
typedef __attribute__((ext_vector_type(8)))  _Float16 v8h;
typedef __attribute__((ext_vector_type(8)))  float    v8f;

// ---------------- Workspace layout (bytes) ----------------
static constexpr size_t AL(size_t x) { return (x + 255) & ~size_t(255); }
static constexpr size_t OFF_META = 0;                                    // counts[8], cursor[8], offsets[8]
static constexpr size_t OFF_TOKE = 1024;                                 // int  [T][2] expert ids (topk order)
static constexpr size_t OFF_TOKW = AL(OFF_TOKE + T_TOK*2*4);             // f32  [T][2] weights
static constexpr size_t OFF_TOKP = AL(OFF_TOKW + T_TOK*2*4);             // int  [T][2] positions in segments
static constexpr size_t OFF_TOKF = AL(OFF_TOKP + T_TOK*2*4);             // int  [2T]   token id per position
static constexpr size_t OFF_NX   = AL(OFF_TOKF + 2*T_TOK*4);             // f16  [T][2560] normalized input
static constexpr size_t OFF_ACTS = AL(OFF_NX   + (size_t)T_TOK*IN_DIM*2);// f16  [T][512]  shared swiglu act
static constexpr size_t OFF_SOUT = AL(OFF_ACTS + (size_t)T_TOK*HID*2);   // f32  [T][2048] shared branch out
static constexpr size_t OFF_ACTE = AL(OFF_SOUT + (size_t)T_TOK*LLM*4);   // f16  [2T][512] expert swiglu act
static constexpr size_t OFF_EO   = AL(OFF_ACTE + (size_t)2*T_TOK*HID*2); // f32  [2T][2048] expert outputs

// =====================================================================
// Kernel 1: fused LN-pre (over 2560) + router logits + softmax + top-2.
// One block (256 thr) per token; each thread owns 10 strided elements.
// =====================================================================
__global__ void ln_router_kernel(const float* __restrict__ x,
                                 const float* __restrict__ gpre,
                                 const float* __restrict__ bpre,
                                 const float* __restrict__ rw,
                                 const float* __restrict__ rb,
                                 h16*  __restrict__ nx,
                                 int*  __restrict__ token_e,
                                 float* __restrict__ token_w,
                                 int*  __restrict__ counts)
{
    const int t   = blockIdx.x;
    const int tid = threadIdx.x;
    const float* xt = x + (size_t)t * IN_DIM;   // token data is contiguous (S even)

    float xr[10];
    float s = 0.f, ss = 0.f;
#pragma unroll
    for (int i = 0; i < 10; ++i) {
        float v = xt[tid + 256*i];
        xr[i] = v; s += v; ss += v*v;
    }
    __shared__ float sbuf[256];
    __shared__ float smean, srstd, lg[NEXP];

    sbuf[tid] = s; __syncthreads();
    for (int off = 128; off > 0; off >>= 1) {
        if (tid < off) sbuf[tid] += sbuf[tid+off];
        __syncthreads();
    }
    if (tid == 0) smean = sbuf[0] * (1.f/IN_DIM);
    __syncthreads();
    sbuf[tid] = ss; __syncthreads();
    for (int off = 128; off > 0; off >>= 1) {
        if (tid < off) sbuf[tid] += sbuf[tid+off];
        __syncthreads();
    }
    if (tid == 0) {
        float m = smean;
        float var = sbuf[0] * (1.f/IN_DIM) - m*m;
        srstd = rsqrtf(var + 1e-6f);
    }
    __syncthreads();

    const float m = smean, rs = srstd;
    float plog[NEXP] = {0,0,0,0,0,0,0,0};
#pragma unroll
    for (int i = 0; i < 10; ++i) {
        int d = tid + 256*i;
        float n = (xr[i] - m) * rs * gpre[d] + bpre[d];
        nx[(size_t)t*IN_DIM + d] = (h16)n;
#pragma unroll
        for (int e = 0; e < NEXP; ++e) plog[e] += n * rw[e*IN_DIM + d];
    }
    for (int e = 0; e < NEXP; ++e) {
        __syncthreads();
        sbuf[tid] = plog[e]; __syncthreads();
        for (int off = 128; off > 0; off >>= 1) {
            if (tid < off) sbuf[tid] += sbuf[tid+off];
            __syncthreads();
        }
        if (tid == 0) lg[e] = sbuf[0] + rb[e];
    }
    __syncthreads();

    if (tid == 0) {
        float mx = lg[0];
        for (int e = 1; e < NEXP; ++e) mx = fmaxf(mx, lg[e]);
        float p[NEXP], sum = 0.f;
        for (int e = 0; e < NEXP; ++e) { p[e] = expf(lg[e] - mx); sum += p[e]; }
        float inv = 1.f / sum;
        for (int e = 0; e < NEXP; ++e) p[e] *= inv;
        int i0 = 0; float b0 = p[0];
        for (int e = 1; e < NEXP; ++e) if (p[e] > b0) { b0 = p[e]; i0 = e; }
        int i1 = (i0 == 0) ? 1 : 0; float b1 = p[i1];
        for (int e = 0; e < NEXP; ++e) if (e != i0 && p[e] > b1) { b1 = p[e]; i1 = e; }
        float denom = 1.f / (b0 + b1 + 1e-20f);
        token_e[t*2+0] = i0; token_e[t*2+1] = i1;
        token_w[t*2+0] = b0 * denom; token_w[t*2+1] = b1 * denom;
        atomicAdd(&counts[i0], 1);
        atomicAdd(&counts[i1], 1);
    }
}

// =====================================================================
// Kernel 2: exclusive prefix over 8 expert counts
// =====================================================================
__global__ void prefix_kernel(int* __restrict__ meta)
{
    if (threadIdx.x == 0 && blockIdx.x == 0) {
        int off = 0;
        for (int e = 0; e < NEXP; ++e) { meta[16 + e] = off; off += meta[e]; }
    }
}

// =====================================================================
// Kernel 3: scatter tokens to compacted per-expert segments.
// Slot order may vary run-to-run but never affects any computed value.
// =====================================================================
__global__ void scatter_kernel(const int* __restrict__ token_e,
                               int* __restrict__ cursor,
                               const int* __restrict__ offsets,
                               int* __restrict__ tokf,
                               int* __restrict__ token_pos)
{
    int t = blockIdx.x * 256 + threadIdx.x;
    if (t >= T_TOK) return;
#pragma unroll
    for (int k = 0; k < 2; ++k) {
        int e    = token_e[t*2 + k];
        int slot = atomicAdd(&cursor[e], 1);
        int pos  = offsets[e] + slot;
        tokf[pos] = t;
        token_pos[t*2 + k] = pos;
    }
}

// ---------------------------------------------------------------------
// Fragment loaders (ISA 7.12.2 layouts).
// A (16x32): lanes<16 hold K{0..7,16..23}, lanes>=16 hold K{8..15,24..31}.
// B (32x16): lane = N column; lanes<16 hold K{0..7,16..23}, lanes>=16 hold K{8..15,24..31}.
// ---------------------------------------------------------------------
__device__ __forceinline__ v16h frag_a(const h16* As, int m_sub, int lane, int ks)
{
    const h16* row = As + (m_sub*16 + (lane & 15)) * 64 + ks;
    const int  kb  = (lane < 16) ? 0 : 8;
    v8h lo = *reinterpret_cast<const v8h*>(row + kb);
    v8h hi = *reinterpret_cast<const v8h*>(row + kb + 16);
    v16h a;
#pragma unroll
    for (int i = 0; i < 8; ++i) { a[i] = lo[i]; a[8+i] = hi[i]; }
    return a;
}
__device__ __forceinline__ v16h frag_b(const h16* Bs, int nrow_base, int lane, int ks)
{
    const h16* row = Bs + (nrow_base + (lane & 15)) * 64 + ks + ((lane < 16) ? 0 : 8);
    v8h lo = *reinterpret_cast<const v8h*>(row);
    v8h hi = *reinterpret_cast<const v8h*>(row + 16);
    v16h b;
#pragma unroll
    for (int i = 0; i < 8; ++i) { b[i] = lo[i]; b[8+i] = hi[i]; }
    return b;
}

// =====================================================================
// WMMA GEMM, BM=64 x BN=128 x BK=64, 256 threads = 8 waves,
// 2x2 register tile per wave (4 accumulators, 8 WMMA per K-chunk).
//
// SWIGLU=true  (first projection): block covers 64 act columns
//   [n0, n0+64). B-panel rows 0..63 = gate weights (rows n0+r of w12),
//   rows 64..127 = value weights (rows 512+n0+r). Wave w owns gate tile
//   g=w>>1 and its matching value tile g+4, so silu(gate)*val is purely
//   register-local in the epilogue; writes f16 act [rows, 512].
// SWIGLU=false (second projection): plain 128 output columns, f32 out.
//
// Grouped-GEMM support: gridDim.z = segment; rows are
//   a_row = GATHER ? tokf[seg_base+m] : seg_base+m, early-exit past count.
// =====================================================================
template<int KDIM, bool GATHER, bool SWIGLU>
__global__ __launch_bounds__(256)
void gemm_wmma(const h16*  __restrict__ A,
               const float* __restrict__ Bw,
               void*        __restrict__ Out,
               const int*   __restrict__ counts,
               const int*   __restrict__ offsets,
               const int*   __restrict__ tokf,
               int Mfixed, int NtotB)
{
    const int z        = blockIdx.z;
    const int seg_base = offsets ? offsets[z] : 0;
    const int m_count  = counts  ? counts[z]  : Mfixed;
    const int m0       = blockIdx.x * 64;
    if (m0 >= m_count) return;
    const int n0       = blockIdx.y * (SWIGLU ? 64 : 128);
    const float* Bseg  = Bw + (size_t)z * NtotB * KDIM;

    __shared__ __align__(16) h16 As[64*64];     // 8 KB
    __shared__ __align__(16) h16 Bs[128*64];    // 16 KB

    const int tid  = threadIdx.x;
    const int lane = tid & 31;
    const int wave = tid >> 5;
    const int m_subA = (wave & 1) * 2;          // m_subs {m_subA, m_subA+1}
    const int npart  = wave >> 1;               // 0..3

    // B-panel tile rows for this wave
    const int brow0 = SWIGLU ? (npart*16)        : (npart*32);
    const int brow1 = SWIGLU ? (64 + npart*16)   : (npart*32 + 16);

    v8f acc00 = {}, acc01 = {}, acc10 = {}, acc11 = {};

    // staging coords
    const int lrowA = tid >> 2;            // 64 rows, 4 thr/row, 16 halves each
    const int lkkA  = (tid & 3) * 16;
    const int lrowB = tid >> 1;            // 128 rows, 2 thr/row, 32 halves each
    const int lkkB  = (tid & 1) * 32;

    for (int k0 = 0; k0 < KDIM; k0 += 64) {
        __syncthreads();
        // ---- stage A panel (f16, 64x64) ----
        {
            const int m_loc = m0 + lrowA;
            uint4 v0 = make_uint4(0u,0u,0u,0u), v1 = v0;
            if (m_loc < m_count) {
                int arow = GATHER ? tokf[seg_base + m_loc] : (seg_base + m_loc);
                const uint4* src = reinterpret_cast<const uint4*>(
                    A + (size_t)arow * KDIM + k0 + lkkA);
                v0 = src[0]; v1 = src[1];
            }
            uint4* dst = reinterpret_cast<uint4*>(&As[lrowA*64 + lkkA]);
            dst[0] = v0; dst[1] = v1;
        }
        // ---- stage B panel (f32 -> f16, 128x64; weights L2-resident) ----
        {
            const int wrow = SWIGLU
                ? ((lrowB < 64) ? (n0 + lrowB) : (HID + n0 + (lrowB - 64)))
                : (n0 + lrowB);
            const float* src = Bseg + (size_t)wrow * KDIM + k0 + lkkB;
            h16 tmp[32];
#pragma unroll
            for (int q = 0; q < 8; ++q) {
                float4 f = reinterpret_cast<const float4*>(src)[q];
                tmp[q*4+0]=(h16)f.x; tmp[q*4+1]=(h16)f.y;
                tmp[q*4+2]=(h16)f.z; tmp[q*4+3]=(h16)f.w;
            }
            h16* dst = &Bs[lrowB*64 + lkkB];
#pragma unroll
            for (int q = 0; q < 4; ++q)
                reinterpret_cast<uint4*>(dst)[q] =
                    reinterpret_cast<const uint4*>(tmp)[q];
        }
        __syncthreads();

#pragma unroll
        for (int ks = 0; ks < 64; ks += 32) {
            v16h a0 = frag_a(As, m_subA,     lane, ks);
            v16h a1 = frag_a(As, m_subA + 1, lane, ks);
            v16h b0 = frag_b(Bs, brow0, lane, ks);
            v16h b1 = frag_b(Bs, brow1, lane, ks);
            acc00 = __builtin_amdgcn_wmma_f32_16x16x32_f16(false, a0, false, b0, (short)0, acc00, false, false);
            acc01 = __builtin_amdgcn_wmma_f32_16x16x32_f16(false, a0, false, b1, (short)0, acc01, false, false);
            acc10 = __builtin_amdgcn_wmma_f32_16x16x32_f16(false, a1, false, b0, (short)0, acc10, false, false);
            acc11 = __builtin_amdgcn_wmma_f32_16x16x32_f16(false, a1, false, b1, (short)0, acc11, false, false);
        }
    }

    // ---- epilogue (D layout: lane<16 -> M=r, lane>=16 -> M=8+r; col=lane&15) ----
    const int hi8 = (lane < 16) ? 0 : 8;
    const int col = lane & 15;
#pragma unroll
    for (int mi = 0; mi < 2; ++mi) {
        const v8f& ag = mi ? acc10 : acc00;
        const v8f& av = mi ? acc11 : acc01;
#pragma unroll
        for (int r = 0; r < 8; ++r) {
            int m_loc = m0 + (m_subA + mi)*16 + hi8 + r;
            if (m_loc < m_count) {
                size_t orow = (size_t)(seg_base + m_loc);
                if (SWIGLU) {
                    float g = ag[r], v = av[r];
                    float sv = (g / (1.f + expf(-g))) * v;
                    ((h16*)Out)[orow * HID + n0 + npart*16 + col] = (h16)sv;
                } else {
                    float* O = (float*)Out;
                    int cA = n0 + npart*32 + col;
                    O[orow * NtotB + cA]      = ag[r];
                    O[orow * NtotB + cA + 16] = av[r];
                }
            }
        }
    }
}

// =====================================================================
// Combine (pull both expert rows in fixed top-k order => deterministic)
// + LN-post over 2048, write final f32 output.
// =====================================================================
__global__ void combine_ln_kernel(const float* __restrict__ sOut,
                                  const float* __restrict__ eo,
                                  const int*   __restrict__ token_pos,
                                  const float* __restrict__ token_w,
                                  const float* __restrict__ gpost,
                                  const float* __restrict__ bpost,
                                  float* __restrict__ out)
{
    const int t   = blockIdx.x;
    const int tid = threadIdx.x;
    const int p0 = token_pos[t*2], p1 = token_pos[t*2+1];
    const float w0 = token_w[t*2], w1 = token_w[t*2+1];

    float vr[8];
    float s = 0.f, ss = 0.f;
#pragma unroll
    for (int i = 0; i < 8; ++i) {
        int d = tid + 256*i;
        float v = sOut[(size_t)t*LLM + d]
                + w0 * eo[(size_t)p0*LLM + d]
                + w1 * eo[(size_t)p1*LLM + d];
        vr[i] = v; s += v; ss += v*v;
    }
    __shared__ float sbuf[256];
    __shared__ float smean, srstd;
    sbuf[tid] = s; __syncthreads();
    for (int off = 128; off > 0; off >>= 1) {
        if (tid < off) sbuf[tid] += sbuf[tid+off];
        __syncthreads();
    }
    if (tid == 0) smean = sbuf[0] * (1.f/LLM);
    __syncthreads();
    sbuf[tid] = ss; __syncthreads();
    for (int off = 128; off > 0; off >>= 1) {
        if (tid < off) sbuf[tid] += sbuf[tid+off];
        __syncthreads();
    }
    if (tid == 0) {
        float m = smean;
        float var = sbuf[0] * (1.f/LLM) - m*m;
        srstd = rsqrtf(var + 1e-6f);
    }
    __syncthreads();
    const float m = smean, rs = srstd;
#pragma unroll
    for (int i = 0; i < 8; ++i) {
        int d = tid + 256*i;
        out[(size_t)t*LLM + d] = (vr[i] - m) * rs * gpost[d] + bpost[d];
    }
}

// =====================================================================
extern "C" void kernel_launch(void* const* d_in, const int* in_sizes, int n_in,
                              void* d_out, int out_size, void* d_ws, size_t ws_size,
                              hipStream_t stream)
{
    const float* x      = (const float*)d_in[0];
    const float* gpre   = (const float*)d_in[1];
    const float* bpre   = (const float*)d_in[2];
    const float* rw     = (const float*)d_in[3];
    const float* rb     = (const float*)d_in[4];
    const float* w12s   = (const float*)d_in[5];
    const float* w3s    = (const float*)d_in[6];
    const float* w12e   = (const float*)d_in[7];
    const float* w3e    = (const float*)d_in[8];
    const float* gpost  = (const float*)d_in[9];
    const float* bpost  = (const float*)d_in[10];
    float* out = (float*)d_out;

    char* ws = (char*)d_ws;
    int*   meta      = (int*)(ws + OFF_META);   // [0..7]=counts [8..15]=cursor [16..23]=offsets
    int*   counts    = meta;
    int*   cursor    = meta + 8;
    int*   offsets   = meta + 16;
    int*   token_e   = (int*)  (ws + OFF_TOKE);
    float* token_w   = (float*)(ws + OFF_TOKW);
    int*   token_pos = (int*)  (ws + OFF_TOKP);
    int*   tokf      = (int*)  (ws + OFF_TOKF);
    h16*   nx        = (h16*)  (ws + OFF_NX);
    h16*   acts      = (h16*)  (ws + OFF_ACTS);
    float* sOut      = (float*)(ws + OFF_SOUT);
    h16*   acte      = (h16*)  (ws + OFF_ACTE);
    float* eo        = (float*)(ws + OFF_EO);

    const int MT = (T_TOK + 63) / 64;  // 188 M-tiles

    hipMemsetAsync(meta, 0, 64, stream);  // counts + cursor = 0

    ln_router_kernel<<<T_TOK, 256, 0, stream>>>(x, gpre, bpre, rw, rb,
                                                nx, token_e, token_w, counts);
    prefix_kernel<<<1, 32, 0, stream>>>(meta);
    scatter_kernel<<<(T_TOK + 255) / 256, 256, 0, stream>>>(token_e, cursor, offsets,
                                                            tokf, token_pos);

    // Shared branch: fused GEMM+SwiGLU -> acts[T,512]; then acts x w3s^T -> sOut[T,2048]
    gemm_wmma<IN_DIM, false, true ><<<dim3(MT, HID/64, 1), 256, 0, stream>>>(
        nx, w12s, acts, nullptr, nullptr, nullptr, T_TOK, HID2);
    gemm_wmma<HID, false, false><<<dim3(MT, LLM/128, 1), 256, 0, stream>>>(
        acts, w3s, sOut, nullptr, nullptr, nullptr, T_TOK, LLM);

    // Expert branch (grouped GEMM over compacted segments, gridDim.z = 8 experts)
    gemm_wmma<IN_DIM, true , true ><<<dim3(MT, HID/64, NEXP), 256, 0, stream>>>(
        nx, w12e, acte, counts, offsets, tokf, 0, HID2);
    gemm_wmma<HID, false, false><<<dim3(MT, LLM/128, NEXP), 256, 0, stream>>>(
        acte, w3e, eo, counts, offsets, nullptr, 0, LLM);

    // Deterministic pull-combine + LN-post
    combine_ln_kernel<<<T_TOK, 256, 0, stream>>>(sOut, eo, token_pos, token_w,
                                                 gpost, bpost, out);
}
